// GCNEncoder_81449759801842
// MI455X (gfx1250) — matrix operand order repaired and checked by reading, not compiled, not run on hardware
//
#include <hip/hip_runtime.h>
#include <hip/hip_bf16.h>
#include <stdint.h>

typedef __attribute__((ext_vector_type(16))) _Float16 v16h;
typedef __attribute__((ext_vector_type(4)))  _Float16 v4h;
typedef __attribute__((ext_vector_type(8)))  float    v8f;

#define N_NODES  50000
#define N_EDGES  600000
#define D        128
#define N_GRAPHS 256
#define BN_EPS   1e-5f

// ---------------------------------------------------------------- utilities
__global__ void fill_f32(float* __restrict__ p, float v, int n) {
    int i = blockIdx.x * blockDim.x + threadIdx.x;
    if (i < n) p[i] = v;
}

// Pack W (128x128 f32, row-major [k][n]) into WMMA B-fragment order:
// Wp[((wave*4 + kb)*32 + lane)*16 + i] = f16( W[(kb*32 + i + 16*(lane>=16)) * 128 + wave*16 + lane%16] )
// so each lane's 16-half B fragment is one contiguous 32-byte load.
__global__ void pack_w_kernel(const float* __restrict__ W, _Float16* __restrict__ Wp) {
    int t = blockIdx.x * blockDim.x + threadIdx.x;
    if (t < D * D) {
        int i    =  t        & 15;
        int lane = (t >> 4)  & 31;
        int kb   = (t >> 9)  & 3;
        int w    = (t >> 11) & 7;
        int k = kb * 32 + i + ((lane >= 16) ? 16 : 0);
        int n = w * 16 + (lane & 15);
        Wp[t] = (_Float16)W[k * D + n];
    }
}

// ------------------------------------------------- degree / normalization
__global__ void deg_kernel(const long long* __restrict__ ei, float* __restrict__ deg) {
    int e = blockIdx.x * blockDim.x + threadIdx.x;
    if (e < N_EDGES) {
        int dst = (int)ei[N_EDGES + e];
        atomicAdd(&deg[dst], 1.0f);
    }
}

__global__ void rsqrt_kernel(float* __restrict__ deg) {
    int i = blockIdx.x * blockDim.x + threadIdx.x;
    if (i < N_NODES) deg[i] = rsqrtf(deg[i]);   // deg >= 1 always (self loop)
}

// ------------------------------------------------------------- WMMA GEMM
// H[m,n] = sum_k X[m,k]*W[k,n] + bias[n].  One block = 16 rows (50000 = 3125*16
// exactly, so no row guards).  8 waves, each owns 16 output columns.
__global__ __launch_bounds__(256) void gemm_bias_wmma(
    const float*    __restrict__ X,
    const v16h*     __restrict__ Wp,     // fragment-packed f16 weights
    const float*    __restrict__ bias,
    float*          __restrict__ H)
{
    __shared__ _Float16 As[16][D + 8];   // +8 halves pad

    const int m0  = blockIdx.x * 16;
    const int tid = threadIdx.x;

    // stage 16x128 A tile: float4 global loads -> f16 -> b64 LDS stores
    const float4* X4 = (const float4*)X;
    for (int idx = tid; idx < 16 * 32; idx += 256) {     // 512 quads, 2 iters
        int r = idx >> 5, k4 = idx & 31;
        float4 v = X4[(size_t)(m0 + r) * 32 + k4];
        v4h h4 = { (_Float16)v.x, (_Float16)v.y, (_Float16)v.z, (_Float16)v.w };
        *(v4h*)&As[r][k4 * 4] = h4;
    }
    __syncthreads();

    const int wave = tid >> 5;
    const int lane = tid & 31;
    const int hi   = lane >> 4;          // half-wave select
    const int ln   = lane & 15;

    __builtin_prefetch(&Wp[(wave * 4) * 32 + lane], 0, 3);   // global_prefetch_b8

    v8f acc = {0.f, 0.f, 0.f, 0.f, 0.f, 0.f, 0.f, 0.f};

#pragma unroll
    for (int kb = 0; kb < 4; ++kb) {     // K = 128 in 4 steps of 32
        v16h a;
#pragma unroll
        for (int i = 0; i < 16; ++i) {
            // A k order used here (same as packed B): lanes 0-15 K 0..15; lanes 16-31 K 16..31
            int ka = kb * 32 + i + hi * 16;
            a[i] = As[ln][ka];           // 2x ds_load_b128 per fragment
        }
        v16h b = Wp[(wave * 4 + kb) * 32 + lane];   // 2x global_load_b128
        acc = __builtin_amdgcn_wmma_f32_16x16x32_f16(
                  false, a, false, b, (short)0, acc, false, false);
    }

    const int col  = wave * 16 + ln;
    const float bc = bias[col];
#pragma unroll
    for (int r = 0; r < 8; ++r) {        // C/D layout: VGPR r -> M = r + 8*hi
        int row = m0 + r + hi * 8;
        H[(size_t)row * D + col] = acc[r] + bc;
    }
}

// --------------------------------------------------------- message passing
// self-loop term written directly (initializes agg, no zeroing pass); float4
__global__ void selfloop_kernel(const float4* __restrict__ Ht,
                                const float*  __restrict__ dinv,
                                float4* __restrict__ agg)
{
    int g = blockIdx.x * blockDim.x + threadIdx.x;        // over N*32 quads
    if (g < N_NODES * (D / 4)) {
        int i = g >> 5;
        float s = dinv[i]; s *= s;
        float4 v = Ht[g];
        v.x *= s; v.y *= s; v.z *= s; v.w *= s;
        agg[g] = v;
    }
}

// one thread per (edge, feature-quad): float4 gather + 4 atomic adds
__global__ void scatter_kernel(const float4* __restrict__ Ht,
                               const float*  __restrict__ dinv,
                               const long long* __restrict__ ei,
                               float* __restrict__ agg)
{
    int g = blockIdx.x * blockDim.x + threadIdx.x;        // over E*32 quads
    if (g < N_EDGES * (D / 4)) {
        int e  = g >> 5;
        int q  = g & 31;                  // feature quad 0..31
        int s  = (int)ei[e];
        int t  = (int)ei[N_EDGES + e];
        float norm = dinv[s] * dinv[t];
        float4 v = Ht[(size_t)s * (D / 4) + q];
        float* dst = &agg[(size_t)t * D + q * 4];
        atomicAdd(dst + 0, v.x * norm);
        atomicAdd(dst + 1, v.y * norm);
        atomicAdd(dst + 2, v.z * norm);
        atomicAdd(dst + 3, v.w * norm);
    }
}

// -------------------------------------------------------------- batch norm
__global__ void bn_stats_kernel(const float4* __restrict__ H,
                                float* __restrict__ stats, int nblocks)
{
    int tid  = threadIdx.x;
    int q    = tid & 31;                  // feature quad
    int sub  = tid >> 5;                  // 0..7
    int row0 = blockIdx.x * 8 + sub;
    int stride = nblocks * 8;
    float4 s  = {0.f, 0.f, 0.f, 0.f};
    float4 s2 = {0.f, 0.f, 0.f, 0.f};
    for (int i = row0; i < N_NODES; i += stride) {
        float4 v = H[(size_t)i * (D / 4) + q];
        s.x += v.x;       s.y += v.y;       s.z += v.z;       s.w += v.w;
        s2.x += v.x * v.x; s2.y += v.y * v.y; s2.z += v.z * v.z; s2.w += v.w * v.w;
    }
    int d = q * 4;
    atomicAdd(&stats[d + 0], s.x);  atomicAdd(&stats[d + 1], s.y);
    atomicAdd(&stats[d + 2], s.z);  atomicAdd(&stats[d + 3], s.w);
    atomicAdd(&stats[D + d + 0], s2.x);  atomicAdd(&stats[D + d + 1], s2.y);
    atomicAdd(&stats[D + d + 2], s2.z);  atomicAdd(&stats[D + d + 3], s2.w);
}

__global__ void bn_relu_kernel(const float4* __restrict__ Hin,
                               const float*  __restrict__ stats,
                               const float*  __restrict__ gamma,
                               const float*  __restrict__ beta,
                               float4* __restrict__ Hout)
{
    int g = blockIdx.x * blockDim.x + threadIdx.x;        // over N*32 quads
    if (g < N_NODES * (D / 4)) {
        int d = (g & 31) * 4;
        const float invN = 1.0f / (float)N_NODES;
        float4 v = Hin[g];
        float4 o;
        {
            float mu = stats[d] * invN;
            float vr = stats[D + d] * invN - mu * mu;
            o.x = fmaxf((v.x - mu) * rsqrtf(vr + BN_EPS) * gamma[d] + beta[d], 0.f);
        }
        {
            float mu = stats[d + 1] * invN;
            float vr = stats[D + d + 1] * invN - mu * mu;
            o.y = fmaxf((v.y - mu) * rsqrtf(vr + BN_EPS) * gamma[d + 1] + beta[d + 1], 0.f);
        }
        {
            float mu = stats[d + 2] * invN;
            float vr = stats[D + d + 2] * invN - mu * mu;
            o.z = fmaxf((v.z - mu) * rsqrtf(vr + BN_EPS) * gamma[d + 2] + beta[d + 2], 0.f);
        }
        {
            float mu = stats[d + 3] * invN;
            float vr = stats[D + d + 3] * invN - mu * mu;
            o.w = fmaxf((v.w - mu) * rsqrtf(vr + BN_EPS) * gamma[d + 3] + beta[d + 3], 0.f);
        }
        Hout[g] = o;
    }
}

// -------------------------------------------------------------------- pool
__global__ void pool_accum_kernel(const float4* __restrict__ H,
                                  const long long* __restrict__ batch,
                                  float* __restrict__ sums,
                                  float* __restrict__ cnt)
{
    int g = blockIdx.x * blockDim.x + threadIdx.x;        // over N*32 quads
    if (g < N_NODES * (D / 4)) {
        int i = g >> 5, q = g & 31;
        int b = (int)batch[i];
        float4 v = H[g];
        float* dst = &sums[(size_t)b * D + q * 4];
        atomicAdd(dst + 0, v.x);
        atomicAdd(dst + 1, v.y);
        atomicAdd(dst + 2, v.z);
        atomicAdd(dst + 3, v.w);
        if (q == 0) atomicAdd(&cnt[b], 1.0f);
    }
}

__global__ void pool_div_kernel(float* __restrict__ out, const float* __restrict__ cnt) {
    int g = blockIdx.x * blockDim.x + threadIdx.x;
    if (g < N_GRAPHS * D) out[g] /= fmaxf(cnt[g >> 7], 1.0f);
}

// ------------------------------------------------------------------ driver
extern "C" void kernel_launch(void* const* d_in, const int* in_sizes, int n_in,
                              void* d_out, int out_size, void* d_ws, size_t ws_size,
                              hipStream_t stream)
{
    const float*     x   = (const float*)d_in[0];
    const float*     W1  = (const float*)d_in[1];
    const float*     b1  = (const float*)d_in[2];
    const float*     g1  = (const float*)d_in[3];
    const float*     be1 = (const float*)d_in[4];
    const float*     W2  = (const float*)d_in[5];
    const float*     b2  = (const float*)d_in[6];
    const float*     g2  = (const float*)d_in[7];
    const float*     be2 = (const float*)d_in[8];
    const long long* ei    = (const long long*)d_in[9];
    const long long* batch = (const long long*)d_in[10];
    float* out = (float*)d_out;

    // workspace carve-up (~51.3 MB)
    char*  ws  = (char*)d_ws;
    size_t off = 0;
    auto carve = [&](size_t bytes) -> void* {
        void* p = ws + off;
        off += (bytes + 255) & ~(size_t)255;
        return p;
    };
    _Float16* W1p   = (_Float16*)carve((size_t)D * D * sizeof(_Float16));
    _Float16* W2p   = (_Float16*)carve((size_t)D * D * sizeof(_Float16));
    float*    dinv  = (float*)carve((size_t)N_NODES * sizeof(float));
    float*    stats = (float*)carve(2 * D * sizeof(float));
    float*    cnt   = (float*)carve(N_GRAPHS * sizeof(float));
    float*    hA    = (float*)carve((size_t)N_NODES * D * sizeof(float));
    float*    hB    = (float*)carve((size_t)N_NODES * D * sizeof(float));

    const int T = 256;
    const int gW    = (D * D + T - 1) / T;                // 64
    const int gN    = (N_NODES + T - 1) / T;              // 196
    const int gE    = (N_EDGES + T - 1) / T;              // 2344
    const int gNQ   = (N_NODES * (D / 4) + T - 1) / T;    // 6250
    const int gEQ   = (N_EDGES * (D / 4) + T - 1) / T;    // 75000
    const int gGemm = N_NODES / 16;                       // 3125 exact
    const int gStat = 512;
    const int gOut  = (N_GRAPHS * D + T - 1) / T;         // 128

    // normalization: deg = 1 (self loop) + in-edge counts; dinv = rsqrt(deg)
    fill_f32<<<gN, T, 0, stream>>>(dinv, 1.0f, N_NODES);
    deg_kernel<<<gE, T, 0, stream>>>(ei, dinv);
    rsqrt_kernel<<<gN, T, 0, stream>>>(dinv);

    // weights -> fragment-packed f16 once per call
    pack_w_kernel<<<gW, T, 0, stream>>>(W1, W1p);
    pack_w_kernel<<<gW, T, 0, stream>>>(W2, W2p);

    // -------- layer 1: x -> hA (GEMM) -> hB (agg) -> hA (BN+ReLU)
    gemm_bias_wmma<<<gGemm, T, 0, stream>>>(x, (const v16h*)W1p, b1, hA);
    selfloop_kernel<<<gNQ, T, 0, stream>>>((const float4*)hA, dinv, (float4*)hB);
    scatter_kernel<<<gEQ, T, 0, stream>>>((const float4*)hA, dinv, ei, hB);
    fill_f32<<<1, T, 0, stream>>>(stats, 0.0f, 2 * D);
    bn_stats_kernel<<<gStat, T, 0, stream>>>((const float4*)hB, stats, gStat);
    bn_relu_kernel<<<gNQ, T, 0, stream>>>((const float4*)hB, stats, g1, be1, (float4*)hA);

    // -------- layer 2: hA -> hB (GEMM) -> hA (agg) -> hB (BN+ReLU)
    gemm_bias_wmma<<<gGemm, T, 0, stream>>>(hA, (const v16h*)W2p, b2, hB);
    selfloop_kernel<<<gNQ, T, 0, stream>>>((const float4*)hB, dinv, (float4*)hA);
    scatter_kernel<<<gEQ, T, 0, stream>>>((const float4*)hB, dinv, ei, hA);
    fill_f32<<<1, T, 0, stream>>>(stats, 0.0f, 2 * D);
    bn_stats_kernel<<<gStat, T, 0, stream>>>((const float4*)hA, stats, gStat);
    bn_relu_kernel<<<gNQ, T, 0, stream>>>((const float4*)hA, stats, g2, be2, (float4*)hB);

    // -------- global mean pool
    fill_f32<<<gOut, T, 0, stream>>>(out, 0.0f, N_GRAPHS * D);
    fill_f32<<<1, T, 0, stream>>>(cnt, 0.0f, N_GRAPHS);
    pool_accum_kernel<<<gNQ, T, 0, stream>>>((const float4*)hB, batch, out, cnt);
    pool_div_kernel<<<gOut, T, 0, stream>>>(out, cnt);
}
